// GCN_64656437674592
// MI455X (gfx1250) — matrix operand order repaired and checked by reading, not compiled, not run on hardware
//
#include <hip/hip_runtime.h>

// ---------------------------------------------------------------------------
// 3-layer GCN for MI455X / gfx1250 (wave32).
//
// GEMMs: V_WMMA_F32_16X16X32_BF16 with fp32 = bf16_hi + bf16_lo split
// (3 WMMAs per K=32 step, fp32 accumulate -> ~2^-15 relative error).
// Split precomputed into global bf16 buffers (A row-padded, W transposed +
// col-padded) so the GEMM is LDS-free: every WMMA fragment is two contiguous
// 16B global loads per lane. Each wave owns a 32x64 output tile so the four
// B-fragment pairs are reused across two M-tiles (24 loads : 24 WMMAs per
// K-step). Aggregation: L2-resident float4 gather + hw fp32 atomic scatter
// (feature table 51MB << 192MB L2).
// ---------------------------------------------------------------------------

typedef __attribute__((ext_vector_type(16))) __bf16 v16bf;
typedef __attribute__((ext_vector_type(8)))  __bf16 v8bf;
typedef __attribute__((ext_vector_type(4)))  __bf16 v4bf;
typedef __attribute__((ext_vector_type(8)))  float  v8f;

#define KDIM 256          // feature dim (K of every GEMM)
#define ROWTILE 256       // rows per workgroup (8 waves x 32 rows)

__device__ inline v16bf cat8(v8bf a, v8bf b) {
  return __builtin_shufflevector(a, b, 0,1,2,3,4,5,6,7,8,9,10,11,12,13,14,15);
}

__device__ inline void atomic_add_f32(float* p, float v) {
  __hip_atomic_fetch_add(p, v, __ATOMIC_RELAXED, __HIP_MEMORY_SCOPE_AGENT);
}

// --------------------------- fp32 -> bf16 hi/lo ----------------------------

// Features: X[N x 256] fp32 -> Hi/Lo[Npad x 256] bf16 (rows >= N zero-filled).
template <bool RELU>
__global__ __launch_bounds__(256) void k_split_feat(
    const float* __restrict__ X, __bf16* __restrict__ Hi, __bf16* __restrict__ Lo,
    int N, int Npad) {
  int idx = blockIdx.x * 256 + threadIdx.x;        // one float4 per thread
  if (idx >= Npad * (KDIM / 4)) return;
  int node = idx >> 6;                             // / (256/4)
  int c = (idx & 63) << 2;
  float4 v = make_float4(0.f, 0.f, 0.f, 0.f);
  if (node < N) {
    v = *(const float4*)(X + (size_t)node * KDIM + c);
    if (RELU) {
      v.x = fmaxf(v.x, 0.f); v.y = fmaxf(v.y, 0.f);
      v.z = fmaxf(v.z, 0.f); v.w = fmaxf(v.w, 0.f);
    }
  }
  v4bf h, l;
  h[0] = (__bf16)v.x; l[0] = (__bf16)(v.x - (float)h[0]);
  h[1] = (__bf16)v.y; l[1] = (__bf16)(v.y - (float)h[1]);
  h[2] = (__bf16)v.z; l[2] = (__bf16)(v.z - (float)h[2]);
  h[3] = (__bf16)v.w; l[3] = (__bf16)(v.w - (float)h[3]);
  size_t off = (size_t)node * KDIM + c;            // 8B aligned
  *(v4bf*)(Hi + off) = h;
  *(v4bf*)(Lo + off) = l;
}

// Weights: W[256 x Dout] fp32 -> Hi/Lo[DoutPad x 256] bf16, TRANSPOSED [col][k],
// cols >= Dout zero-filled.
__global__ __launch_bounds__(256) void k_split_wt(
    const float* __restrict__ W, __bf16* __restrict__ Hi, __bf16* __restrict__ Lo,
    int Dout, int DoutPad) {
  int idx = blockIdx.x * 256 + threadIdx.x;        // linear over DoutPad*K
  if (idx >= DoutPad * KDIM) return;
  int c = idx >> 8;                                // col
  int k = idx & 255;
  float v = (c < Dout) ? W[(size_t)k * Dout + c] : 0.f;
  __bf16 h = (__bf16)v;
  Hi[idx] = h;                                     // layout [c][k] == linear idx
  Lo[idx] = (__bf16)(v - (float)h);
}

// ------------------------------ WMMA GEMM ----------------------------------
// C[N x Dout] = A[N x 256] @ W[256 x Dout]; A from padded bf16 hi/lo,
// B from transposed padded bf16 hi/lo. LDS-free; each wave: 32x64 tile.
__global__ __launch_bounds__(256) void gcn_gemm_wmma(
    const __bf16* __restrict__ Ahi, const __bf16* __restrict__ Alo,
    const __bf16* __restrict__ Bhi, const __bf16* __restrict__ Blo,  // [col][K]
    float* __restrict__ C, int N, int Dout) {
  const int wave   = threadIdx.x >> 5;
  const int lane   = threadIdx.x & 31;
  const int mrow   = lane & 15;        // M%16 (A) / N%16 (B)
  const int hiHalf = lane >> 4;
  const int kA     = hiHalf * 8;       // A-fragment K offset (ISA 7.12.2)
  const int kB     = hiHalf * 8;       // B-fragment K offset
  const int row0   = (blockIdx.x * 8 + wave) * 32; // 32-row wave tile
  const int col0   = blockIdx.y * 64;

  const __bf16* aH0 = Ahi + (size_t)(row0 + mrow) * KDIM + kA;
  const __bf16* aL0 = Alo + (size_t)(row0 + mrow) * KDIM + kA;
  const __bf16* aH1 = aH0 + (size_t)16 * KDIM;
  const __bf16* aL1 = aL0 + (size_t)16 * KDIM;

  v8f acc[2][4] = {};

  for (int kk = 0; kk < KDIM; kk += 32) {
    __builtin_prefetch(aH0 + kk + 64, 0, 1);       // global_prefetch_b8
    v16bf ahi0 = cat8(*(const v8bf*)(aH0 + kk), *(const v8bf*)(aH0 + kk + 16));
    v16bf alo0 = cat8(*(const v8bf*)(aL0 + kk), *(const v8bf*)(aL0 + kk + 16));
    v16bf ahi1 = cat8(*(const v8bf*)(aH1 + kk), *(const v8bf*)(aH1 + kk + 16));
    v16bf alo1 = cat8(*(const v8bf*)(aL1 + kk), *(const v8bf*)(aL1 + kk + 16));
    #pragma unroll
    for (int n = 0; n < 4; ++n) {
      const __bf16* bH = Bhi + (size_t)(col0 + n * 16 + mrow) * KDIM + kk + kB;
      const __bf16* bL = Blo + (size_t)(col0 + n * 16 + mrow) * KDIM + kk + kB;
      v16bf bhi = cat8(*(const v8bf*)bH, *(const v8bf*)(bH + 16));
      v16bf blo = cat8(*(const v8bf*)bL, *(const v8bf*)(bL + 16));
      // acc += Ahi*Bhi + Ahi*Blo + Alo*Bhi  for both M-tiles (B reused)
      acc[0][n] = __builtin_amdgcn_wmma_f32_16x16x32_bf16(false, ahi0, false, bhi,
                                                          (short)0, acc[0][n], false, false);
      acc[0][n] = __builtin_amdgcn_wmma_f32_16x16x32_bf16(false, ahi0, false, blo,
                                                          (short)0, acc[0][n], false, false);
      acc[0][n] = __builtin_amdgcn_wmma_f32_16x16x32_bf16(false, alo0, false, bhi,
                                                          (short)0, acc[0][n], false, false);
      acc[1][n] = __builtin_amdgcn_wmma_f32_16x16x32_bf16(false, ahi1, false, bhi,
                                                          (short)0, acc[1][n], false, false);
      acc[1][n] = __builtin_amdgcn_wmma_f32_16x16x32_bf16(false, ahi1, false, blo,
                                                          (short)0, acc[1][n], false, false);
      acc[1][n] = __builtin_amdgcn_wmma_f32_16x16x32_bf16(false, alo1, false, bhi,
                                                          (short)0, acc[1][n], false, false);
    }
  }

  // D layout: VGPR r -> row r (lanes 0-15) / r+8 (lanes 16-31), col = lane%16.
  #pragma unroll
  for (int m = 0; m < 2; ++m) {
    const int mbase = row0 + m * 16 + hiHalf * 8;
    #pragma unroll
    for (int n = 0; n < 4; ++n) {
      int gc = col0 + n * 16 + mrow;
      if (gc < Dout) {
        #pragma unroll
        for (int r = 0; r < 8; ++r) {
          int gr = mbase + r;
          if (gr < N) C[(size_t)gr * Dout + gc] = acc[m][n][r];
        }
      }
    }
  }
}

// ----------------------------- aggregation ---------------------------------

__global__ void k_deg_zero(float* deg, int n) {
  int i = blockIdx.x * blockDim.x + threadIdx.x;
  if (i < n) deg[i] = 0.0f;
}

__global__ void k_deg_count(const int* __restrict__ dst, float* deg, int E) {
  int e = blockIdx.x * blockDim.x + threadIdx.x;
  if (e < E) atomic_add_f32(&deg[dst[e]], 1.0f);
}

__global__ void k_deg_rsqrt(float* deg, int n) {   // dinv = rsqrt(deg + 1)
  int i = blockIdx.x * blockDim.x + threadIdx.x;
  if (i < n) deg[i] = rsqrtf(deg[i] + 1.0f);
}

// out[i,:] = hw[i,:]*dinv[i]^2 + bias  (self-loop + bias; full overwrite)
template <int D>
__global__ __launch_bounds__(256) void k_agg_init(
    const float* __restrict__ hw, const float* __restrict__ dinv,
    const float* __restrict__ bias, float* __restrict__ out, int N) {
  constexpr int per = D >> 2;
  int idx = blockIdx.x * 256 + threadIdx.x;
  if (idx >= N * per) return;
  int node = idx / per;
  int c = (idx % per) << 2;
  float s = dinv[node]; s *= s;
  const float4 v = *(const float4*)(hw + (size_t)node * D + c);
  float4 o;
  o.x = v.x * s + bias[c + 0];
  o.y = v.y * s + bias[c + 1];
  o.z = v.z * s + bias[c + 2];
  o.w = v.w * s + bias[c + 3];
  *(float4*)(out + (size_t)node * D + c) = o;
}

// out[dst,:] += hw[src,:] * dinv[src]*dinv[dst]
template <int D>
__global__ __launch_bounds__(256) void k_agg_scatter(
    const float* __restrict__ hw, const float* __restrict__ dinv,
    const int* __restrict__ src, const int* __restrict__ dst,
    float* __restrict__ out, int E) {
  constexpr int per = D >> 2;
  int idx = blockIdx.x * 256 + threadIdx.x;
  if (idx >= E * per) return;            // <= 51.2M, fits int32
  int e = idx / per;
  int c = (idx % per) << 2;
  int s = src[e], d = dst[e];
  float nrm = dinv[s] * dinv[d];
  const float4 v = *(const float4*)(hw + (size_t)s * D + c);
  float* o = out + (size_t)d * D + c;
  atomic_add_f32(o + 0, v.x * nrm);
  atomic_add_f32(o + 1, v.y * nrm);
  atomic_add_f32(o + 2, v.z * nrm);
  atomic_add_f32(o + 3, v.w * nrm);
}

// ------------------------------- driver ------------------------------------

extern "C" void kernel_launch(void* const* d_in, const int* in_sizes, int n_in,
                              void* d_out, int out_size, void* d_ws, size_t ws_size,
                              hipStream_t stream) {
  const float* x  = (const float*)d_in[0];
  const int*   ei = (const int*)d_in[1];
  const float* W1 = (const float*)d_in[2];
  const float* b1 = (const float*)d_in[3];
  const float* Wm = (const float*)d_in[4];
  const float* bm = (const float*)d_in[5];
  const float* W2 = (const float*)d_in[6];
  const float* b2 = (const float*)d_in[7];
  float* out = (float*)d_out;

  const int N    = in_sizes[0] / KDIM;             // 50000
  const int E    = in_sizes[1] / 2;                // 800000
  const int DO   = in_sizes[7];                    // 40
  const int Npad = ((N + ROWTILE - 1) / ROWTILE) * ROWTILE;
  const int DOpad = 64;

  const int* src = ei;
  const int* dst = ei + E;

  // ws: dinv[N] | h[N*256]f32 | hw[N*256]f32 | Ahi/Alo[Npad*256]bf16 | Wt hi/lo
  char* ws = (char*)d_ws;
  float* dinv = (float*)ws;
  size_t off = (((size_t)N * 4) + 255) & ~(size_t)255;
  float* h  = (float*)(ws + off);        off += (size_t)N * KDIM * 4;
  float* hw = (float*)(ws + off);        off += (size_t)N * KDIM * 4;
  __bf16* Ahi = (__bf16*)(ws + off);     off += (size_t)Npad * KDIM * 2;
  __bf16* Alo = (__bf16*)(ws + off);     off += (size_t)Npad * KDIM * 2;
  __bf16* Whi = (__bf16*)(ws + off);     off += (size_t)KDIM * KDIM * 2;
  __bf16* Wlo = (__bf16*)(ws + off);

  const int T = 256;
  const int featBlocks = (Npad * (KDIM / 4) + T - 1) / T;
  const int wtBlocksW  = (KDIM * KDIM + T - 1) / T;       // wide layers
  const int wtBlocksL  = (DOpad * KDIM + T - 1) / T;      // last layer
  dim3 gWide((Npad / ROWTILE), KDIM / 64);
  dim3 gLast((Npad / ROWTILE), DOpad / 64);

  // degrees -> dinv
  k_deg_zero <<<(N + T - 1) / T, T, 0, stream>>>(dinv, N);
  k_deg_count<<<(E + T - 1) / T, T, 0, stream>>>(dst, dinv, E);
  k_deg_rsqrt<<<(N + T - 1) / T, T, 0, stream>>>(dinv, N);

  // ---- layer 1
  k_split_feat<false><<<featBlocks, T, 0, stream>>>(x, Ahi, Alo, N, Npad);
  k_split_wt<<<wtBlocksW, T, 0, stream>>>(W1, Whi, Wlo, KDIM, KDIM);
  gcn_gemm_wmma<<<gWide, T, 0, stream>>>(Ahi, Alo, Whi, Wlo, hw, N, KDIM);
  k_agg_init<KDIM>   <<<(N * 64 + T - 1) / T, T, 0, stream>>>(hw, dinv, b1, h, N);
  k_agg_scatter<KDIM><<<(E * 64 + T - 1) / T, T, 0, stream>>>(hw, dinv, src, dst, h, E);

  // ---- layer 2 (relu fused into split)
  k_split_feat<true><<<featBlocks, T, 0, stream>>>(h, Ahi, Alo, N, Npad);
  k_split_wt<<<wtBlocksW, T, 0, stream>>>(Wm, Whi, Wlo, KDIM, KDIM);
  gcn_gemm_wmma<<<gWide, T, 0, stream>>>(Ahi, Alo, Whi, Wlo, hw, N, KDIM);
  k_agg_init<KDIM>   <<<(N * 64 + T - 1) / T, T, 0, stream>>>(hw, dinv, bm, h, N);
  k_agg_scatter<KDIM><<<(E * 64 + T - 1) / T, T, 0, stream>>>(hw, dinv, src, dst, h, E);

  // ---- layer 3 (narrow; no final relu)
  k_split_feat<true><<<featBlocks, T, 0, stream>>>(h, Ahi, Alo, N, Npad);
  k_split_wt<<<wtBlocksL, T, 0, stream>>>(W2, Whi, Wlo, DO, DOpad);
  gcn_gemm_wmma<<<gLast, T, 0, stream>>>(Ahi, Alo, Whi, Wlo, hw, N, DO);
  k_agg_init<40>   <<<(N * 10 + T - 1) / T, T, 0, stream>>>(hw, dinv, b2, out, N);
  k_agg_scatter<40><<<(E * 10 + T - 1) / T, T, 0, stream>>>(hw, dinv, src, dst, out, E);
}
